// ViT_10969346474560
// MI455X (gfx1250) — matrix operand order repaired and checked by reading, not compiled, not run on hardware
//
#include <hip/hip_runtime.h>

// ---- problem constants (ViT single block, B=64) ----
#define BATCH   64
#define S_TOK   197
#define S_PAD   224          // 14 tiles of 16, 7 K-chunks of 32
#define DMODEL  768
#define NHEAD   12
#define DHEAD   64
#define NPATCH  196
#define OUTC    1000
#define OUTP    1024         // padded to 16 groups of 64 columns

typedef __attribute__((ext_vector_type(16))) _Float16 v16h;
typedef __attribute__((ext_vector_type(8)))  _Float16 v8h;
typedef __attribute__((ext_vector_type(8)))  float    v8f;

// ---------------- WMMA fragment helpers (ISA 7.12.2 layouts, wave32) ---------
__device__ __forceinline__ v16h cat16(v8h lo, v8h hi) {
  return __builtin_shufflevector(lo, hi, 0, 1, 2, 3, 4, 5, 6, 7,
                                 8, 9, 10, 11, 12, 13, 14, 15);
}
// A (16x32 f16) row-major, 16B-aligned rows: lane l -> row m=l&15;
// lanes<16 hold K[0..7],[16..23]; lanes>=16 hold K[8..15],[24..31]
__device__ __forceinline__ v16h load_a_frag(const _Float16* A, int lda) {
  const int l  = threadIdx.x & 31;
  const int m  = l & 15;
  const int kk = (l < 16) ? 0 : 8;
  const _Float16* p = A + (size_t)m * lda + kk;
  v8h lo = *(const v8h*)p;          // K kk..kk+7      (global_load_b128)
  v8h hi = *(const v8h*)(p + 16);   // K 16+kk..23+kk  (global_load_b128)
  return cat16(lo, hi);
}
// B fragment from an N-major ("transposed") operand BT[n][k], 16B-aligned rows:
// lane l -> col n=l&15; lanes<16 hold K[0..7],[16..23]; lanes>=16 hold K[8..15],[24..31]
__device__ __forceinline__ v16h load_bt_frag(const _Float16* BT, int ldk) {
  const int l  = threadIdx.x & 31;
  const int n  = l & 15;
  const int kb = (l < 16) ? 0 : 8;
  const _Float16* p = BT + (size_t)n * ldk + kb;
  v8h lo = *(const v8h*)p;
  v8h hi = *(const v8h*)(p + 16);
  return cat16(lo, hi);
}
__device__ __forceinline__ v8f wmma_f16(v16h a, v16h b, v8f c) {
  return __builtin_amdgcn_wmma_f32_16x16x32_f16(false, a, false, b, (short)0, c,
                                                false, false);
}
__device__ __forceinline__ v8f vzero8() {
  v8f z = {0.f, 0.f, 0.f, 0.f, 0.f, 0.f, 0.f, 0.f};
  return z;
}

// ---------------- converts / transposes --------------------------------------
__global__ void f32_to_f16_kernel(const float* __restrict__ in,
                                  _Float16* __restrict__ out, size_t n) {
  size_t i = (size_t)blockIdx.x * blockDim.x + threadIdx.x;
  if (i < n) out[i] = (_Float16)in[i];
}
// W[K,N] f32 -> WT[N,K] f16
__global__ void transpose_f32_to_f16_kernel(const float* __restrict__ w,
                                            _Float16* __restrict__ out,
                                            int K, int N) {
  size_t i = (size_t)blockIdx.x * blockDim.x + threadIdx.x;
  if (i >= (size_t)K * N) return;
  int n = (int)(i / K), k = (int)(i % K);
  out[i] = (_Float16)w[(size_t)k * N + n];
}
// wq/wk/wv [NH, DH(d), DH(e)] f32 -> [NH, DH(e), DH(d)] f16
__global__ void transpose_heads_kernel(const float* __restrict__ w,
                                       _Float16* __restrict__ out) {
  size_t i = (size_t)blockIdx.x * blockDim.x + threadIdx.x;
  if (i >= (size_t)NHEAD * DHEAD * DHEAD) return;
  int h = (int)(i / (DHEAD * DHEAD));
  int r = (int)(i % (DHEAD * DHEAD));
  int e = r / DHEAD, d = r % DHEAD;
  out[i] = (_Float16)w[(size_t)h * DHEAD * DHEAD + (size_t)d * DHEAD + e];
}
// w_head [768,1000] f32 -> [1024(n),768(k)] f16, zero padded rows
__global__ void whead_padT_kernel(const float* __restrict__ w,
                                  _Float16* __restrict__ out) {
  size_t i = (size_t)blockIdx.x * blockDim.x + threadIdx.x;
  if (i >= (size_t)OUTP * DMODEL) return;
  int n = (int)(i / DMODEL), k = (int)(i % DMODEL);
  out[i] = (n < OUTC) ? (_Float16)w[(size_t)k * OUTC + n] : (_Float16)0.f;
}

// -------- generic WMMA GEMM: C = act(A@B + bias) + resid, B given N-major ----
// wave computes 16x64 (4 col-tiles sharing one A fragment). M%16==0, N%64==0,
// K%32==0. Tile guards are wave-uniform => EXEC all-ones around WMMA.
__global__ void gemm_wmma_kernel(const _Float16* __restrict__ A,
                                 const _Float16* __restrict__ BT,
                                 const float* __restrict__ bias,
                                 const float* __restrict__ resid,
                                 float* __restrict__ C,
                                 int M, int N, int K,
                                 int lda, int ldk, int ldc, int relu) {
  const int ng    = N >> 6;
  const int wave  = blockIdx.x * (blockDim.x >> 5) + (threadIdx.x >> 5);
  const int total = (M >> 4) * ng;
  if (wave >= total) return;
  const int tm = wave / ng, tg = wave % ng;
  const _Float16* Ab = A + (size_t)(tm * 16) * lda;
  const _Float16* Bb = BT + (size_t)(tg * 64) * ldk;
  v8f acc[4];
#pragma unroll
  for (int t = 0; t < 4; ++t) acc[t] = vzero8();
  for (int k = 0; k < K; k += 32) {
    if (k + 32 < K) __builtin_prefetch(Ab + k + 32, 0, 3);
    v16h a = load_a_frag(Ab + k, lda);
#pragma unroll
    for (int t = 0; t < 4; ++t) {
      v16h b = load_bt_frag(Bb + (size_t)(t * 16) * ldk + k, ldk);
      acc[t] = wmma_f16(a, b, acc[t]);
    }
  }
  const int l  = threadIdx.x & 31;
  const int rb = tm * 16 + ((l < 16) ? 0 : 8);
#pragma unroll
  for (int t = 0; t < 4; ++t) {
    const int n = tg * 64 + t * 16 + (l & 15);
    float bb = bias ? bias[n] : 0.f;
#pragma unroll
    for (int r = 0; r < 8; ++r) {
      float v = acc[t][r] + bb;
      if (relu) v = fmaxf(v, 0.f);
      size_t idx = (size_t)(rb + r) * ldc + n;
      if (resid) v += resid[idx];
      C[idx] = v;
    }
  }
}

// ---------------- assemble tokens: cls + patch-proj + bias + sinusoidal PE ---
__global__ void assemble_tokens_kernel(const float* __restrict__ tmp,
                                       const float* __restrict__ b_map,
                                       const float* __restrict__ cls_tok,
                                       float* __restrict__ tokens) {
  size_t i = (size_t)blockIdx.x * blockDim.x + threadIdx.x;
  if (i >= (size_t)BATCH * S_TOK * DMODEL) return;
  int d = (int)(i % DMODEL);
  size_t bs = i / DMODEL;
  int s = (int)(bs % S_TOK), b = (int)(bs / S_TOK);
  float jj   = (d & 1) ? (float)(d - 1) : (float)d;
  float freq = __powf(10000.f, jj / (float)DMODEL);
  float ang  = (float)s / freq;
  float pe   = (d & 1) ? __cosf(ang) : __sinf(ang);
  float v = (s == 0)
                ? cls_tok[d]
                : tmp[((size_t)b * NPATCH + (s - 1)) * DMODEL + d] + b_map[d];
  tokens[i] = v + pe;
}

// ---------------- LayerNorm over (S,D) per sample, emits f16 -----------------
__global__ void ln2d_kernel(const float* __restrict__ x,
                            const float* __restrict__ g,
                            const float* __restrict__ bta,
                            _Float16* __restrict__ out) {
  const int b = blockIdx.x;
  const size_t N = (size_t)S_TOK * DMODEL;
  const float* xb = x + (size_t)b * N;
  __shared__ float s_sum[256], s_sq[256];
  float sum = 0.f, sq = 0.f;
  for (size_t i = threadIdx.x; i < N; i += blockDim.x) {
    float v = xb[i]; sum += v; sq += v * v;
  }
  s_sum[threadIdx.x] = sum; s_sq[threadIdx.x] = sq;
  __syncthreads();
  for (int o = 128; o > 0; o >>= 1) {
    if ((int)threadIdx.x < o) {
      s_sum[threadIdx.x] += s_sum[threadIdx.x + o];
      s_sq[threadIdx.x]  += s_sq[threadIdx.x + o];
    }
    __syncthreads();
  }
  float mu  = s_sum[0] / (float)N;
  float var = s_sq[0] / (float)N - mu * mu;
  float ri  = rsqrtf(var + 1e-5f);
  for (size_t i = threadIdx.x; i < N; i += blockDim.x) {
    out[(size_t)b * N + i] = (_Float16)((xb[i] - mu) * ri * g[i] + bta[i]);
  }
}

// ---------------- QKV projection: wave computes one 16-row slab of one head --
// grid (197, NHEAD, 3=q/k/v), block 128 (4 waves = 4 consecutive row-tiles)
__global__ void qkv_kernel(const _Float16* __restrict__ xln,   // [B*S, D]
                           const _Float16* __restrict__ wqkvT, // [3,NH,DH(e),DH(d)]
                           const float* __restrict__ bq,
                           const float* __restrict__ bk,
                           const float* __restrict__ bv,
                           _Float16* __restrict__ q,    // [B,NH,S_PAD,DH]
                           _Float16* __restrict__ kk,   // [B,NH,S_PAD,DH]
                           _Float16* __restrict__ vT) { // [B,NH,DH,S_PAD]
  const int rt  = blockIdx.x * 4 + (threadIdx.x >> 5);
  const int h   = blockIdx.y, mat = blockIdx.z;
  const _Float16* WT = wqkvT + ((size_t)mat * NHEAD + h) * DHEAD * DHEAD;
  const _Float16* Ab = xln + (size_t)(rt * 16) * DMODEL + h * DHEAD;
  v8f acc[4];
#pragma unroll
  for (int t = 0; t < 4; ++t) acc[t] = vzero8();
#pragma unroll
  for (int k = 0; k < DHEAD; k += 32) {
    v16h a = load_a_frag(Ab + k, DMODEL);
#pragma unroll
    for (int t = 0; t < 4; ++t)
      acc[t] = wmma_f16(a, load_bt_frag(WT + (size_t)(t * 16) * DHEAD + k, DHEAD),
                        acc[t]);
  }
  const int l  = threadIdx.x & 31;
  const int rb = rt * 16 + ((l < 16) ? 0 : 8);
  const float* bias = (mat == 0) ? bq : (mat == 1) ? bk : bv;
#pragma unroll
  for (int t = 0; t < 4; ++t) {
    const int n = t * 16 + (l & 15);
    const float bb = bias[h * DHEAD + n];
#pragma unroll
    for (int r = 0; r < 8; ++r) {
      int row = rb + r;
      if (row >= BATCH * S_TOK) continue;
      int b_ = row / S_TOK, s = row % S_TOK;
      _Float16 val = (_Float16)(acc[t][r] + bb);
      size_t slab = (size_t)b_ * NHEAD + h;
      if (mat == 0)      q [(slab * S_PAD + s) * DHEAD + n] = val;
      else if (mat == 1) kk[(slab * S_PAD + s) * DHEAD + n] = val;
      else               vT[(slab * DHEAD + n) * S_PAD + s] = val;
    }
  }
}

// ---------------- fused attention per (b,h,16-row tile), one wave ------------
__global__ void attn_kernel(const _Float16* __restrict__ q,
                            const _Float16* __restrict__ kk,
                            const _Float16* __restrict__ vT,
                            const float* __restrict__ tokens,
                            float* __restrict__ out) {
  __shared__ alignas(16) _Float16 qs[16 * DHEAD];   // 2KB q tile
  __shared__ float                sc[16 * S_PAD];   // scores
  __shared__ alignas(16) _Float16 af[16 * S_PAD];   // attn weights f16
  const int idx   = blockIdx.x;
  const int stile = idx % (S_PAD / 16);
  const int bh    = idx / (S_PAD / 16);
  const int h = bh % NHEAD, b = bh / NHEAD;
  const _Float16* qb = q  + ((size_t)bh * S_PAD + stile * 16) * DHEAD;
  const _Float16* kb = kk + (size_t)bh * S_PAD * DHEAD;
  const _Float16* vb = vT + (size_t)bh * DHEAD * S_PAD;
  const int l = threadIdx.x & 31;

  // --- async-stage the contiguous 2KB q tile into LDS (ASYNCcnt path) ---
  {
    unsigned lbase = (unsigned)(uintptr_t)(&qs[0]);
    unsigned long long gbase = (unsigned long long)(uintptr_t)qb;
#pragma unroll
    for (int i = 0; i < 4; ++i) {
      unsigned ldst = lbase + (unsigned)(i * 512 + l * 16);
      unsigned long long ga = gbase + (unsigned long long)(i * 512 + l * 16);
      asm volatile("global_load_async_to_lds_b128 %0, %1, off"
                   :: "v"(ldst), "v"(ga) : "memory");
    }
    asm volatile("s_wait_asynccnt 0x0" ::: "memory");
  }
  __syncthreads();

  v16h a0 = load_a_frag(qs, DHEAD);
  v16h a1 = load_a_frag(qs + 32, DHEAD);
  // scores = q @ k^T / 8   (k stored row-major [s][d] == N-major operand)
  for (int nt = 0; nt < S_PAD / 16; ++nt) {
    v8f acc = vzero8();
    acc = wmma_f16(a0, load_bt_frag(kb + (size_t)(nt * 16) * DHEAD + 0, DHEAD), acc);
    acc = wmma_f16(a1, load_bt_frag(kb + (size_t)(nt * 16) * DHEAD + 32, DHEAD), acc);
    const int n  = nt * 16 + (l & 15);
    const int rb = (l < 16) ? 0 : 8;
#pragma unroll
    for (int r = 0; r < 8; ++r) sc[(rb + r) * S_PAD + n] = acc[r] * 0.125f;
  }
  __syncthreads();
  // masked row softmax (32 lanes cooperate per row)
  for (int m = 0; m < 16; ++m) {
    float mx = -1e30f;
    for (int c = l; c < S_PAD; c += 32)
      mx = fmaxf(mx, (c < S_TOK) ? sc[m * S_PAD + c] : -1e30f);
#pragma unroll
    for (int o = 16; o > 0; o >>= 1) mx = fmaxf(mx, __shfl_xor(mx, o, 32));
    float sm = 0.f;
    for (int c = l; c < S_PAD; c += 32) {
      float e = (c < S_TOK) ? __expf(sc[m * S_PAD + c] - mx) : 0.f;
      sc[m * S_PAD + c] = e;
      sm += e;
    }
#pragma unroll
    for (int o = 16; o > 0; o >>= 1) sm += __shfl_xor(sm, o, 32);
    float rs = 1.f / sm;
    for (int c = l; c < S_PAD; c += 32)
      af[m * S_PAD + c] = (_Float16)(sc[m * S_PAD + c] * rs);
  }
  __syncthreads();
  // o = attn @ v   (M=16, K=S_PAD, N=64); v stored transposed [d][s] == N-major
  v8f oac[4];
#pragma unroll
  for (int t = 0; t < 4; ++t) oac[t] = vzero8();
  for (int kc = 0; kc < S_PAD; kc += 32) {
    v16h a = load_a_frag(af + kc, S_PAD);
#pragma unroll
    for (int t = 0; t < 4; ++t)
      oac[t] = wmma_f16(a, load_bt_frag(vb + (size_t)(t * 16) * S_PAD + kc, S_PAD),
                        oac[t]);
  }
  const int n0 = l & 15;
  const int rb = (l < 16) ? 0 : 8;
#pragma unroll
  for (int t = 0; t < 4; ++t) {
#pragma unroll
    for (int r = 0; r < 8; ++r) {
      int s = stile * 16 + rb + r;
      if (s < S_TOK) {
        size_t oi = ((size_t)b * S_TOK + s) * DMODEL + h * DHEAD + t * 16 + n0;
        out[oi] = tokens[oi] + oac[t][r];
      }
    }
  }
}

// ---------------- extract cls rows as f16 ------------------------------------
__global__ void cls_extract_kernel(const float* __restrict__ out2,
                                   _Float16* __restrict__ cls16) {
  int i = blockIdx.x * blockDim.x + threadIdx.x;
  if (i < BATCH * DMODEL) {
    int b = i / DMODEL, d = i % DMODEL;
    cls16[i] = (_Float16)out2[(size_t)b * S_TOK * DMODEL + d];
  }
}

// ---------------- classifier softmax (adds bias, drops padding) --------------
__global__ void head_softmax_kernel(const float* __restrict__ logits,
                                    const float* __restrict__ b_head,
                                    float* __restrict__ outp) {
  const int b = blockIdx.x;
  __shared__ float red[256];
  float mx = -1e30f;
  for (int c = threadIdx.x; c < OUTC; c += 256)
    mx = fmaxf(mx, logits[(size_t)b * OUTP + c] + b_head[c]);
  red[threadIdx.x] = mx; __syncthreads();
  for (int o = 128; o > 0; o >>= 1) {
    if ((int)threadIdx.x < o)
      red[threadIdx.x] = fmaxf(red[threadIdx.x], red[threadIdx.x + o]);
    __syncthreads();
  }
  mx = red[0]; __syncthreads();
  float sm = 0.f;
  for (int c = threadIdx.x; c < OUTC; c += 256)
    sm += __expf(logits[(size_t)b * OUTP + c] + b_head[c] - mx);
  red[threadIdx.x] = sm; __syncthreads();
  for (int o = 128; o > 0; o >>= 1) {
    if ((int)threadIdx.x < o) red[threadIdx.x] += red[threadIdx.x + o];
    __syncthreads();
  }
  float rs = 1.f / red[0];
  for (int c = threadIdx.x; c < OUTC; c += 256)
    outp[(size_t)b * OUTC + c] =
        __expf(logits[(size_t)b * OUTP + c] + b_head[c] - mx) * rs;
}

// =============================================================================
extern "C" void kernel_launch(void* const* d_in, const int* in_sizes, int n_in,
                              void* d_out, int out_size, void* d_ws,
                              size_t ws_size, hipStream_t stream) {
  const float* images  = (const float*)d_in[0];
  const float* w_map   = (const float*)d_in[1];
  const float* b_map   = (const float*)d_in[2];
  const float* cls_tok = (const float*)d_in[3];
  const float* ln1_g   = (const float*)d_in[4];
  const float* ln1_b   = (const float*)d_in[5];
  const float* wq      = (const float*)d_in[6];
  const float* bq      = (const float*)d_in[7];
  const float* wk      = (const float*)d_in[8];
  const float* bk      = (const float*)d_in[9];
  const float* wv      = (const float*)d_in[10];
  const float* bv      = (const float*)d_in[11];
  const float* ln2_g   = (const float*)d_in[12];
  const float* ln2_b   = (const float*)d_in[13];
  const float* w_enc   = (const float*)d_in[14];
  const float* b_enc   = (const float*)d_in[15];
  const float* w_head  = (const float*)d_in[16];
  const float* b_head  = (const float*)d_in[17];
  float* outp = (float*)d_out;

  // ---- workspace carve-up ----
  char* ws = (char*)d_ws;
  size_t off = 0;
  auto alloc = [&](size_t bytes) {
    size_t o = off;
    off += (bytes + 255) & ~(size_t)255;
    return o;
  };
  const size_t nPatchElems = (size_t)BATCH * NPATCH * DMODEL;
  const size_t nTokElems   = (size_t)BATCH * S_TOK * DMODEL;
  const size_t nQKV        = (size_t)BATCH * NHEAD * S_PAD * DHEAD;

  _Float16* A16     = (_Float16*)(ws + alloc(nPatchElems * 2));
  _Float16* WmapT   = (_Float16*)(ws + alloc((size_t)DMODEL * DMODEL * 2));
  _Float16* WqkvT   = (_Float16*)(ws + alloc((size_t)3 * NHEAD * DHEAD * DHEAD * 2));
  _Float16* WencT   = (_Float16*)(ws + alloc((size_t)DMODEL * DMODEL * 2));
  _Float16* WheadT  = (_Float16*)(ws + alloc((size_t)OUTP * DMODEL * 2));
  float*    tmp     = (float*)(ws + alloc(nPatchElems * 4));
  float*    tokens  = (float*)(ws + alloc(nTokElems * 4));
  _Float16* xln16   = (_Float16*)(ws + alloc(nTokElems * 2));
  _Float16* qbuf    = (_Float16*)(ws + alloc(3 * nQKV * 2));
  _Float16* kbuf    = qbuf + nQKV;
  _Float16* vTbuf   = qbuf + 2 * nQKV;
  float*    outb    = (float*)(ws + alloc(nTokElems * 4));
  _Float16* y16     = (_Float16*)(ws + alloc(nTokElems * 2));
  float*    out2    = (float*)(ws + alloc(nTokElems * 4));
  _Float16* cls16   = (_Float16*)(ws + alloc((size_t)BATCH * DMODEL * 2));
  float*    logits  = (float*)(ws + alloc((size_t)BATCH * OUTP * 4));

  auto cdiv = [](size_t a, size_t b) { return (int)((a + b - 1) / b); };

  // 1) converts / weight transposes
  f32_to_f16_kernel<<<cdiv(nPatchElems, 256), 256, 0, stream>>>(images, A16,
                                                                nPatchElems);
  transpose_f32_to_f16_kernel<<<cdiv((size_t)DMODEL * DMODEL, 256), 256, 0,
                                stream>>>(w_map, WmapT, DMODEL, DMODEL);
  const size_t nWh = (size_t)NHEAD * DHEAD * DHEAD;
  transpose_heads_kernel<<<cdiv(nWh, 256), 256, 0, stream>>>(wq, WqkvT);
  transpose_heads_kernel<<<cdiv(nWh, 256), 256, 0, stream>>>(wk, WqkvT + nWh);
  transpose_heads_kernel<<<cdiv(nWh, 256), 256, 0, stream>>>(wv, WqkvT + 2 * nWh);
  transpose_f32_to_f16_kernel<<<cdiv((size_t)DMODEL * DMODEL, 256), 256, 0,
                                stream>>>(w_enc, WencT, DMODEL, DMODEL);
  whead_padT_kernel<<<cdiv((size_t)OUTP * DMODEL, 256), 256, 0, stream>>>(w_head,
                                                                          WheadT);

  // 2) patch embedding GEMM: [12544,768] x [768,768]
  {
    int M = BATCH * NPATCH, N = DMODEL, K = DMODEL;
    int waves = (M / 16) * (N / 64);
    gemm_wmma_kernel<<<cdiv(waves, 8), 256, 0, stream>>>(
        A16, WmapT, nullptr, nullptr, tmp, M, N, K, DMODEL, DMODEL, DMODEL, 0);
  }
  // 3) cls + bias + positional embedding
  assemble_tokens_kernel<<<cdiv(nTokElems, 256), 256, 0, stream>>>(tmp, b_map,
                                                                   cls_tok, tokens);
  // 4) LN1 -> f16
  ln2d_kernel<<<BATCH, 256, 0, stream>>>(tokens, ln1_g, ln1_b, xln16);

  // 5) QKV projections
  {
    dim3 g((BATCH * S_TOK) / 64, NHEAD, 3);   // 197 x 12 x 3, 4 row-tiles/block
    qkv_kernel<<<g, 128, 0, stream>>>(xln16, WqkvT, bq, bk, bv, qbuf, kbuf, vTbuf);
  }
  // 6) fused attention + residual
  attn_kernel<<<BATCH * NHEAD * (S_PAD / 16), 32, 0, stream>>>(qbuf, kbuf, vTbuf,
                                                               tokens, outb);
  // 7) LN2 -> f16
  ln2d_kernel<<<BATCH, 256, 0, stream>>>(outb, ln2_g, ln2_b, y16);

  // 8) MLP GEMM + bias + ReLU + residual
  {
    int M = BATCH * S_TOK, N = DMODEL, K = DMODEL;
    int waves = (M / 16) * (N / 64);
    gemm_wmma_kernel<<<cdiv(waves, 8), 256, 0, stream>>>(
        y16, WencT, b_enc, outb, out2, M, N, K, DMODEL, DMODEL, DMODEL, 1);
  }
  // 9) classifier head on cls token
  cls_extract_kernel<<<cdiv((size_t)BATCH * DMODEL, 256), 256, 0, stream>>>(out2,
                                                                            cls16);
  {
    int M = BATCH, N = OUTP, K = DMODEL;
    int waves = (M / 16) * (N / 64);
    gemm_wmma_kernel<<<cdiv(waves, 8), 256, 0, stream>>>(
        cls16, WheadT, nullptr, nullptr, logits, M, N, K, DMODEL, DMODEL, OUTP, 0);
  }
  head_softmax_kernel<<<BATCH, 256, 0, stream>>>(logits, b_head, outp);
}
